// Scaled_Dot_Product_Attention_15702400434586
// MI455X (gfx1250) — matrix operand order repaired and checked by reading, not compiled, not run on hardware
//
#include <hip/hip_runtime.h>
#include <hip/hip_bf16.h>

#define SEQ 2048
#define DK  64
#define NBH 32          // B*H
#define NELEM (NBH * SEQ * DK)   // 4,194,304

typedef _Float16 v16h __attribute__((ext_vector_type(16)));
typedef _Float16 v8h  __attribute__((ext_vector_type(8)));
typedef float    v8f  __attribute__((ext_vector_type(8)));

template <bool M> struct BoolC { static constexpr bool value = M; };

__device__ __forceinline__ v8f zero8() {
    v8f z;
#pragma unroll
    for (int i = 0; i < 8; ++i) z[i] = 0.0f;
    return z;
}

__device__ __forceinline__ v8f wmma_f16(v16h a, v16h b, v8f c) {
    // D = A(16x32 f16) * B(32x16 f16) + C(16x16 f32)
    return __builtin_amdgcn_wmma_f32_16x16x32_f16(
        /*neg_a=*/false, a, /*neg_b=*/false, b,
        /*c_mod=*/(short)0, c, /*reuse_a=*/false, /*reuse_b=*/false);
}

// lane l <-> lane l^16 half-wave swap: one VALU op (v_permlanex16_b32).
__device__ __forceinline__ unsigned xor16u(unsigned v) {
    return (unsigned)__builtin_amdgcn_permlanex16(
        (int)v, (int)v, 0x76543210, (int)0xfedcba98u, false, false);
}
__device__ __forceinline__ float xor16f(float v) {
    union { float f; unsigned u; } a, r;
    a.f = v; r.u = xor16u(a.u);
    return r.f;
}

__device__ __forceinline__ unsigned pack2h(float a, float b) {
    union { _Float16 h[2]; unsigned u; } x;
    x.h[0] = (_Float16)a; x.h[1] = (_Float16)b;
    return x.u;
}

__device__ __forceinline__ v16h load_a16(const _Float16* p) {
    // A-operand per-lane pattern: 8 halfs at p[0..7], 8 halfs at p[16..23]
    v8h x = *(const v8h*)p;
    v8h y = *(const v8h*)(p + 16);
    return __builtin_shufflevector(x, y, 0,1,2,3,4,5,6,7,8,9,10,11,12,13,14,15);
}

// ---------------- pre-pass: K f32 -> f16 row-major ----------------
__global__ void cvt_k_kernel(const float* __restrict__ K, _Float16* __restrict__ Kh) {
    int i = blockIdx.x * blockDim.x + threadIdx.x;
    if (i < NELEM) Kh[i] = (_Float16)K[i];
}

// ---------------- pre-pass: V f32 [bh][s][d] -> Vt f16 [bh][d][s] ----------------
__global__ void tr_v_kernel(const float* __restrict__ V, _Float16* __restrict__ Vt) {
    int i = blockIdx.x * blockDim.x + threadIdx.x;   // output-linear: contiguous writes
    if (i >= NELEM) return;
    int s  = i & (SEQ - 1);
    int t  = i >> 11;          // /SEQ
    int d  = t & (DK - 1);
    int bh = t >> 6;           // /DK
    Vt[i] = (_Float16)V[((size_t)bh * SEQ + s) * DK + d];
}

// ---------------- flash attention, one 16-query tile per wave ----------------
__global__ __launch_bounds__(256) void fa_kernel(const float* __restrict__ Q,
                                                 const _Float16* __restrict__ Kh,
                                                 const _Float16* __restrict__ Vt,
                                                 float* __restrict__ O) {
    const int lane = threadIdx.x & 31;
    // Wave index is uniform across the wave; readfirstlane makes it *provably*
    // scalar, so bh/q0/nfull live in SGPRs: the KV loop becomes a scalar
    // s_cbranch loop (no divergent-loop EXEC bookkeeping) and K/V loads use
    // SGPR-base + lane-constant-VGPR-offset addressing.
    const int gw = __builtin_amdgcn_readfirstlane(blockIdx.x * 8 + (threadIdx.x >> 5));
    const int bh = gw >> 7;                     // 128 q-tiles per head
    const int q0 = (127 - (gw & 127)) << 4;     // heavy tiles first (causal balance)
    const bool hi  = lane >= 16;
    const int  ql  = lane & 15;
    const int  aoff = hi ? 8 : 0;               // A-operand lane K-offset

    const _Float16* Kb = Kh + (size_t)bh * SEQ * DK;   // scalar base
    const _Float16* Vb = Vt + (size_t)bh * DK * SEQ;   // scalar base

    // lane-constant (loop-invariant) vector offsets
    const int klane0 = ql * DK + aoff;                 // K rows kv0..kv0+15
    const int klane1 = (16 + ql) * DK + aoff;          // K rows kv0+16..kv0+31

    // ---- hoist B-operand of gemm1 (Q^T, lane = q column).
    // Fold 1/sqrt(64) AND log2(e) into Q so softmax runs in the exp2 domain.
    v16h bq0, bq1;
    {
        const float qs = 0.125f * 1.44269504088896340736f;
        const float* qp = Q + ((size_t)bh * SEQ + q0 + ql) * DK + (hi ? 8 : 0);
#pragma unroll
        for (int i = 0; i < 16; ++i) {
            bq0[i] = (_Float16)(qp[i < 8 ? i : i + 8]      * qs);   // d = c*32 + 8*hi + (i<8 ? i : i+8)
            bq1[i] = (_Float16)(qp[32 + (i < 8 ? i : i + 8)] * qs);
        }
    }

    v8f acc[4] = { zero8(), zero8(), zero8(), zero8() };   // O^T, 4 d-chunks of 16
    float m = -__builtin_inff();
    float l = 0.0f;

    const int qi     = q0 + ql;
    const int nfull  = q0 >> 5;                 // unmasked 32-kv tiles (scalar)
    const int kvlast = nfull << 5;              // diagonal (masked) tile start

    auto do_tile = [&](int kv0, auto MC, bool pf) {
        constexpr bool MASK = decltype(MC)::value;

        if (pf) {   // prefetch next K/V tiles while this one computes
            __builtin_prefetch(Kb + (size_t)(kv0 + 32) * DK + klane0, 0, 3);
            __builtin_prefetch(Vb + (size_t)ql * SEQ + kv0 + 32, 0, 3);
        }

        // ---- gemm1: S^T tile (32 kv x 16 q), c0 = kv 0..15, c1 = kv 16..31
        v8f c0 = zero8(), c1 = zero8();
        {
            const _Float16* kt = Kb + (size_t)kv0 * DK;   // scalar part
            v16h a00 = load_a16(kt + klane0), a01 = load_a16(kt + klane0 + 32);
            v16h a10 = load_a16(kt + klane1), a11 = load_a16(kt + klane1 + 32);
            c0 = wmma_f16(a00, bq0, c0);        // pair same-B back-to-back
            c1 = wmma_f16(a10, bq0, c1);
            c0 = wmma_f16(a01, bq1, c0);
            c1 = wmma_f16(a11, bq1, c1);
        }

        // ---- online softmax (lane: column q = ql; kv rows split lane<->lane^16)
        float s0[8], s1[8];
        float mt = -__builtin_inff();
        const int kvb = kv0 + (hi ? 8 : 0);
#pragma unroll
        for (int r = 0; r < 8; ++r) {
            if (MASK) {
                s0[r] = (kvb + r      <= qi) ? c0[r] : -__builtin_inff();
                s1[r] = (kvb + 16 + r <= qi) ? c1[r] : -__builtin_inff();
            } else {
                s0[r] = c0[r];
                s1[r] = c1[r];
            }
            mt = fmaxf(mt, fmaxf(s0[r], s1[r]));
        }
        mt = fmaxf(mt, xor16f(mt));

        // Running max is monotone: for most tiles no lane raises it, so skip the
        // whole rescale with a wave-uniform branch (EXEC untouched -> WMMA legal).
        if (__any(mt > m)) {
            const float mn = fmaxf(m, mt);
            const float sc = __builtin_amdgcn_exp2f(m - mn);
            l *= sc;
#pragma unroll
            for (int dc = 0; dc < 4; ++dc)
#pragma unroll
                for (int i = 0; i < 8; ++i) acc[dc][i] *= sc;
            m = mn;
        }

        float p0[8], p1[8], rs = 0.0f;
#pragma unroll
        for (int r = 0; r < 8; ++r) {
            p0[r] = __builtin_amdgcn_exp2f(s0[r] - m);
            p1[r] = __builtin_amdgcn_exp2f(s1[r] - m);
            rs += p0[r] + p1[r];
        }
        rs += xor16f(rs);
        l += rs;

        // ---- build B-operand (P^T, 32kv x 16q) in-register: lane-local packing, no swap
        unsigned hp0[4], hp1[4];
#pragma unroll
        for (int j = 0; j < 4; ++j) {
            hp0[j] = pack2h(p0[2 * j], p0[2 * j + 1]);
            hp1[j] = pack2h(p1[2 * j], p1[2 * j + 1]);
        }
        union { unsigned u[8]; v16h h; } B;
#pragma unroll
        for (int j = 0; j < 4; ++j) {
            unsigned o0 = xor16u(hp0[j]);
            unsigned o1 = xor16u(hp1[j]);
            B.u[j]     = hp0[j];   // lo: kv 0..7   | hi: kv 8..15
            B.u[4 + j] = hp1[j];   // lo: kv 16..23 | hi: kv 24..31
        }

        // ---- gemm2: O^T += V^T(16d x 32kv) * P^T(32kv x 16q)
#pragma unroll
        for (int dc = 0; dc < 4; ++dc) {
            const _Float16* vp = Vb + kv0 + (size_t)(dc * 16 + ql) * SEQ + aoff;
            acc[dc] = wmma_f16(load_a16(vp), B.h, acc[dc]);
        }
    };

    for (int t = 0; t < nfull; ++t)
        do_tile(t << 5, BoolC<false>{}, /*prefetch=*/true);
    do_tile(kvlast, BoolC<true>{}, /*prefetch=*/false);   // diagonal tile

    // ---- epilogue: O[q][d] = acc / l   (lane=q, vgpr r -> d = dc*16 + 8*hi + r)
    const float inv = 1.0f / l;
    float* Op = O + ((size_t)bh * SEQ + q0 + ql) * DK + (hi ? 8 : 0);
#pragma unroll
    for (int dc = 0; dc < 4; ++dc) {
        float4 f0, f1;
        f0.x = acc[dc][0] * inv; f0.y = acc[dc][1] * inv;
        f0.z = acc[dc][2] * inv; f0.w = acc[dc][3] * inv;
        f1.x = acc[dc][4] * inv; f1.y = acc[dc][5] * inv;
        f1.z = acc[dc][6] * inv; f1.w = acc[dc][7] * inv;
        *(float4*)(Op + dc * 16)     = f0;
        *(float4*)(Op + dc * 16 + 4) = f1;
    }
}

extern "C" void kernel_launch(void* const* d_in, const int* in_sizes, int n_in,
                              void* d_out, int out_size, void* d_ws, size_t ws_size,
                              hipStream_t stream) {
    const float* Q = (const float*)d_in[0];
    const float* K = (const float*)d_in[1];
    const float* V = (const float*)d_in[2];
    // d_in[3] is the causal mask; causality is applied analytically in-kernel.

    _Float16* Kh = (_Float16*)d_ws;            // 8 MB
    _Float16* Vt = Kh + NELEM;                 // 8 MB  (needs ws_size >= 16 MB)

    const int thr = 256;
    const int blk = (NELEM + thr - 1) / thr;
    cvt_k_kernel<<<blk, thr, 0, stream>>>(K, Kh);
    tr_v_kernel <<<blk, thr, 0, stream>>>(V, Vt);

    // 4096 wave-tiles (32 heads x 128 q-tiles), 8 waves per 256-thread block
    fa_kernel<<<512, 256, 0, stream>>>(Q, Kh, Vt, (float*)d_out);
}
